// GATConv_18657337934239
// MI455X (gfx1250) — matrix operand order repaired and checked by reading, not compiled, not run on hardware
//
#include <hip/hip_runtime.h>

typedef _Float16 v16h __attribute__((ext_vector_type(16)));
typedef float    v8f  __attribute__((ext_vector_type(8)));

#define N_OP_C  20000
#define N_DEV_C 256
#define D_C     128
#define DE_C    64
#define HF_C    512

// ---------------- small prep kernels ----------------

// out[n*K + k] = (f16) W[k*Ncols + n]   (column-major f16 weight for WMMA B operand)
__global__ void transpose_to_f16(const float* __restrict__ W, _Float16* __restrict__ out,
                                 int K, int Ncols) {
    int i = blockIdx.x * blockDim.x + threadIdx.x;
    if (i >= K * Ncols) return;
    int n = i / K, k = i % K;
    out[i] = (_Float16)W[(size_t)k * Ncols + n];
}

// out[k*4 + h] = sum_f W[k*512 + h*128 + f] * a[h*128 + f]
__global__ void fold_attn(const float* __restrict__ W, const float* __restrict__ a,
                          float* __restrict__ out, int K) {
    int i = blockIdx.x * blockDim.x + threadIdx.x;
    if (i >= K * 4) return;
    int k = i / 4, h = i % 4;
    float s = 0.f;
    #pragma unroll 8
    for (int f = 0; f < 128; ++f)
        s += W[(size_t)k * HF_C + h * 128 + f] * a[h * 128 + f];
    out[i] = s;
}

__global__ void fill_f32(float* __restrict__ p, float v, int n) {
    int i = blockIdx.x * blockDim.x + threadIdx.x;
    if (i < n) p[i] = v;
}

// er[n*4+h] = sum_k dstf[n*128+k] * wr[k*4+h]
__global__ void compute_er(const float* __restrict__ dstf, const float* __restrict__ wr,
                           float* __restrict__ er, int Nd) {
    int i = blockIdx.x * blockDim.x + threadIdx.x;
    if (i >= Nd * 4) return;
    int n = i / 4, h = i % 4;
    float s = 0.f;
    #pragma unroll 8
    for (int k = 0; k < 128; ++k)
        s += dstf[(size_t)n * 128 + k] * wr[k * 4 + h];
    er[i] = s;
}

__device__ __forceinline__ void atomicMaxFloat(float* addr, float value) {
    if (value >= 0.f) atomicMax((int*)addr, __float_as_int(value));
    else              atomicMin((unsigned int*)addr, __float_as_uint(value));
}

// one wave per edge: el[h] = x . wl[:,h]; e = leaky_relu(el + er[dst]); seg-max
__global__ void __launch_bounds__(256)
edge_logits_k(const float* __restrict__ srcf, const float* __restrict__ ef,
              const int* __restrict__ src, const int* __restrict__ dst,
              const float* __restrict__ wl, const float* __restrict__ er,
              float* __restrict__ ebuf, float* __restrict__ m, int E) {
    const int lane = threadIdx.x & 31;
    const int e = blockIdx.x * 8 + (threadIdx.x >> 5);
    if (e >= E) return;
    const int s = src[e];
    float a0 = 0.f, a1 = 0.f, a2 = 0.f, a3 = 0.f;
    #pragma unroll
    for (int c = 0; c < 6; ++c) {
        int k = lane + 32 * c;
        float x = (k < D_C) ? srcf[(size_t)s * D_C + k] : ef[(size_t)e * DE_C + (k - D_C)];
        a0 += x * wl[k * 4 + 0];
        a1 += x * wl[k * 4 + 1];
        a2 += x * wl[k * 4 + 2];
        a3 += x * wl[k * 4 + 3];
    }
    #pragma unroll
    for (int off = 16; off > 0; off >>= 1) {
        a0 += __shfl_down(a0, off, 32);
        a1 += __shfl_down(a1, off, 32);
        a2 += __shfl_down(a2, off, 32);
        a3 += __shfl_down(a3, off, 32);
    }
    if (lane == 0) {
        const int d = dst[e];
        float v[4] = {a0, a1, a2, a3};
        #pragma unroll
        for (int h = 0; h < 4; ++h) {
            float t = v[h] + er[d * 4 + h];
            t = (t > 0.f) ? t : 0.2f * t;
            ebuf[(size_t)e * 4 + h] = t;
            atomicMaxFloat(&m[d * 4 + h], t);
        }
    }
}

// p = exp(e - m[dst]); ebuf <- p; z[dst] += p
__global__ void softmax_p_k(const int* __restrict__ dst, float* __restrict__ ebuf,
                            const float* __restrict__ m, float* __restrict__ z, int E) {
    int i = blockIdx.x * blockDim.x + threadIdx.x;
    if (i >= E * 4) return;
    int e = i >> 2, h = i & 3;
    int d = dst[e];
    float p = __expf(ebuf[i] - m[d * 4 + h]);
    ebuf[i] = p;
    atomicAdd(&z[d * 4 + h], p);
}

// ---------------- WMMA pass 3: s-tile recompute + alpha-weighted scatter ----------------
// 16 edges per block, 8 waves, wave w covers column-tiles {w, w+8, w+16, w+24} of 512 cols.
__global__ void __launch_bounds__(256)
edge_agg_wmma(const float* __restrict__ srcf, const float* __restrict__ ef,
              const int* __restrict__ src, const int* __restrict__ dst,
              const _Float16* __restrict__ WT, const float* __restrict__ pbuf,
              const float* __restrict__ z, float* __restrict__ acc, int E) {
    __shared__ _Float16 Xl[16 * 192];
    __shared__ float alpha_s[16 * 4];
    __shared__ int dst_s[16];
    const int tid = threadIdx.x;
    const int e0 = blockIdx.x * 16;

    for (int i = tid; i < 16 * 192; i += 256) {
        int row = i / 192, k = i % 192;
        int e = e0 + row;
        float x = 0.f;
        if (e < E) x = (k < D_C) ? srcf[(size_t)src[e] * D_C + k]
                                 : ef[(size_t)e * DE_C + (k - D_C)];
        Xl[i] = (_Float16)x;
    }
    if (tid < 64) {
        int row = tid >> 2, h = tid & 3;
        int e = e0 + row;
        float a = 0.f;
        if (e < E) { int d = dst[e]; a = pbuf[(size_t)e * 4 + h] / z[d * 4 + h]; }
        alpha_s[tid] = a;
    }
    if (tid < 16) dst_s[tid] = (e0 + tid < E) ? dst[e0 + tid] : 0;
    __syncthreads();

    const int lane = tid & 31, wv = tid >> 5;
    const int rowA = lane & 15;
    const int hv   = lane >> 4;   // half-wave id
    #pragma unroll
    for (int j = 0; j < 4; ++j) {
        const int n0 = (wv + 8 * j) * 16;
        v8f c = {};
        #pragma unroll
        for (int ks = 0; ks < 6; ++ks) {
            const int k0 = ks * 32;
            v16h a, b;
            const _Float16* ap = &Xl[rowA * 192 + k0 + hv * 16];
            #pragma unroll
            for (int i = 0; i < 8; ++i) { a[i] = ap[i]; a[8 + i] = ap[8 + i]; }
            b = *(const v16h*)(WT + (size_t)(n0 + rowA) * 192 + k0 + hv * 16);
            c = __builtin_amdgcn_wmma_f32_16x16x32_f16(false, a, false, b,
                                                       (short)0, c, false, false);
        }
        const int h = n0 >> 7;          // head id (tile fully inside one 128-col head)
        const int ncol = n0 + rowA;
        #pragma unroll
        for (int r = 0; r < 8; ++r) {
            const int row = r + hv * 8;
            atomicAdd(&acc[(size_t)dst_s[row] * HF_C + ncol], c[r] * alpha_s[row * 4 + h]);
        }
    }
}

// ---------------- WMMA finalize: out = feats + (acc*scale) @ W + b ----------------
__global__ void __launch_bounds__(256)
finalize_wmma_k(const float* __restrict__ feats, const float* __restrict__ accv,
                const _Float16* __restrict__ WT, const float* __restrict__ bias,
                float* __restrict__ out, int M, float scale) {
    __shared__ _Float16 Al[16 * 512];
    const int tid = threadIdx.x;
    const int m0 = blockIdx.x * 16;
    for (int i = tid; i < 16 * 512; i += 256) {
        int row = i >> 9, k = i & 511;
        int mm = m0 + row;
        float x = (mm < M) ? accv[(size_t)mm * HF_C + k] * scale : 0.f;
        Al[i] = (_Float16)x;
    }
    __syncthreads();

    const int lane = tid & 31, wv = tid >> 5;
    const int rowA = lane & 15, hv = lane >> 4;
    const int n0 = wv * 16;
    v8f c = {};
    #pragma unroll
    for (int ks = 0; ks < 16; ++ks) {
        const int k0 = ks * 32;
        v16h a, b;
        const _Float16* ap = &Al[rowA * 512 + k0 + hv * 16];
        #pragma unroll
        for (int i = 0; i < 8; ++i) { a[i] = ap[i]; a[8 + i] = ap[8 + i]; }
        b = *(const v16h*)(WT + (size_t)(n0 + rowA) * 512 + k0 + hv * 16);
        c = __builtin_amdgcn_wmma_f32_16x16x32_f16(false, a, false, b,
                                                   (short)0, c, false, false);
    }
    const int ncol = n0 + rowA;
    const float bv = bias[ncol];
    #pragma unroll
    for (int r = 0; r < 8; ++r) {
        const int row = r + hv * 8;
        const int mm = m0 + row;
        if (mm < M)
            out[(size_t)mm * D_C + ncol] = feats[(size_t)mm * D_C + ncol] + bv + c[r];
    }
}

// ---------------- host orchestration ----------------

extern "C" void kernel_launch(void* const* d_in, const int* in_sizes, int n_in,
                              void* d_out, int out_size, void* d_ws, size_t ws_size,
                              hipStream_t stream) {
    const float* op_feats  = (const float*)d_in[0];
    const float* dev_feats = (const float*)d_in[1];
    const float* Wop  = (const float*)d_in[37];
    const float* bop  = (const float*)d_in[38];
    const float* Wdev = (const float*)d_in[39];
    const float* bdev = (const float*)d_in[40];

    char* ws = (char*)d_ws;
    size_t off = 0;
    auto take = [&](size_t bytes) -> char* {
        char* p = ws + off;
        off = (off + bytes + 255) & ~(size_t)255;
        return p;
    };
    _Float16* wsrcT[5];
    for (int t = 0; t < 5; ++t) wsrcT[t] = (_Float16*)take((size_t)192 * 512 * 2);
    _Float16* wopT  = (_Float16*)take((size_t)512 * 128 * 2);
    _Float16* wdevT = (_Float16*)take((size_t)512 * 128 * 2);
    float* wl[5]; for (int t = 0; t < 5; ++t) wl[t] = (float*)take(192 * 4 * 4);
    float* wr[5]; for (int t = 0; t < 5; ++t) wr[t] = (float*)take(128 * 4 * 4);
    float* er   = (float*)take((size_t)N_OP_C * 4 * 4);
    float* mbuf = (float*)take((size_t)N_OP_C * 4 * 4);
    float* zbuf = (float*)take((size_t)N_OP_C * 4 * 4);
    float* ebuf = (float*)take((size_t)160000 * 4 * 4);
    float* acc_op  = (float*)take((size_t)N_OP_C * HF_C * 4);
    float* acc_dev = (float*)take((size_t)N_DEV_C * HF_C * 4);
    (void)ws_size; (void)n_in; (void)out_size;

    // weight prep (deterministic, recomputed each call)
    for (int t = 0; t < 5; ++t) {
        const float* W  = (const float*)d_in[17 + 4 * t];
        const float* Wd = (const float*)d_in[18 + 4 * t];
        const float* al = (const float*)d_in[19 + 4 * t];
        const float* ar = (const float*)d_in[20 + 4 * t];
        transpose_to_f16<<<(192 * 512 + 255) / 256, 256, 0, stream>>>(W, wsrcT[t], 192, 512);
        fold_attn<<<(192 * 4 + 255) / 256, 256, 0, stream>>>(W, al, wl[t], 192);
        fold_attn<<<(128 * 4 + 255) / 256, 256, 0, stream>>>(Wd, ar, wr[t], 128);
    }
    transpose_to_f16<<<(512 * 128 + 255) / 256, 256, 0, stream>>>(Wop, wopT, 512, 128);
    transpose_to_f16<<<(512 * 128 + 255) / 256, 256, 0, stream>>>(Wdev, wdevT, 512, 128);

    fill_f32<<<(N_OP_C * HF_C + 255) / 256, 256, 0, stream>>>(acc_op, 0.f, N_OP_C * HF_C);
    fill_f32<<<(N_DEV_C * HF_C + 255) / 256, 256, 0, stream>>>(acc_dev, 0.f, N_DEV_C * HF_C);

    const int srcIdx[5] = {7, 9, 11, 13, 15};
    const int dstIdx[5] = {8, 10, 12, 14, 16};
    const int efIdx[5]  = {2, 3, 4, 5, 6};
    const int srcDev[5] = {0, 0, 1, 1, 0};
    const int dstDev[5] = {0, 0, 0, 1, 1};
    const float NEG_INF = -__builtin_inff();

    for (int t = 0; t < 5; ++t) {
        const int E  = in_sizes[srcIdx[t]];
        const int Nd = dstDev[t] ? N_DEV_C : N_OP_C;
        const float* sfeat = srcDev[t] ? dev_feats : op_feats;
        const float* dfeat = dstDev[t] ? dev_feats : op_feats;
        const int* src = (const int*)d_in[srcIdx[t]];
        const int* dst = (const int*)d_in[dstIdx[t]];
        const float* ef = (const float*)d_in[efIdx[t]];
        float* acc = dstDev[t] ? acc_dev : acc_op;

        fill_f32<<<(Nd * 4 + 255) / 256, 256, 0, stream>>>(mbuf, NEG_INF, Nd * 4);
        fill_f32<<<(Nd * 4 + 255) / 256, 256, 0, stream>>>(zbuf, 0.f, Nd * 4);
        compute_er<<<(Nd * 4 + 255) / 256, 256, 0, stream>>>(dfeat, wr[t], er, Nd);
        edge_logits_k<<<(E + 7) / 8, 256, 0, stream>>>(sfeat, ef, src, dst, wl[t], er,
                                                       ebuf, mbuf, E);
        softmax_p_k<<<(E * 4 + 255) / 256, 256, 0, stream>>>(dst, ebuf, mbuf, zbuf, E);
        edge_agg_wmma<<<(E + 15) / 16, 256, 0, stream>>>(sfeat, ef, src, dst, wsrcT[t],
                                                         ebuf, zbuf, acc, E);
    }

    finalize_wmma_k<<<(N_OP_C + 15) / 16, 256, 0, stream>>>(
        op_feats, acc_op, wopT, bop, (float*)d_out, N_OP_C, 1.f / 3.f);
    finalize_wmma_k<<<(N_DEV_C + 15) / 16, 256, 0, stream>>>(
        dev_feats, acc_dev, wdevT, bdev, (float*)d_out + (size_t)N_OP_C * D_C, N_DEV_C, 0.5f);
}
